// DGCNNEncoder_11209864643060
// MI455X (gfx1250) — compile-verified, matrix operand order checked by reading, not run on hardware
//
#include <hip/hip_runtime.h>
#include <hip/hip_bf16.h>

typedef __attribute__((ext_vector_type(16))) __bf16 v16bf;
typedef __attribute__((ext_vector_type(8)))  float  v8f;

#define BATCH 16
#define NPTS  2048
#define KNN   20
#define EPSBN 1e-5f
#define SLOPE 0.2f
#define TNPTS 4              // points per edge-conv workgroup -> 80 rows = 5 m-tiles
#define C5ROWS 32            // rows per conv5 workgroup

// ---------------------------------------------------------------- helpers
__device__ inline void atomicMaxFloat(float* addr, float val) {
    // sign-split trick: works for mixed-sign values, init to -inf
    if (val >= 0.0f) atomicMax((int*)addr, __float_as_int(val));
    else             atomicMin((unsigned int*)addr, __float_as_uint(val));
}

// A fragment (16x32 bf16, row-major in LDS [rows][Kpad]) per ISA 7.12.2 layout
__device__ inline v16bf load_a_frag(const __bf16* A, int Kpad, int row0, int k0, int lane) {
    const int m    = lane & 15;
    const int half = lane >> 4;
    const __bf16* rp = A + (size_t)(row0 + m) * Kpad + k0 + half * 8;
    v16bf a;
#pragma unroll
    for (int v = 0; v < 8; ++v) {
        const int k = (v >> 2) * 16 + (v & 3) * 2;   // + half*8 already in rp
        a[2 * v]     = rp[k];
        a[2 * v + 1] = rp[k + 1];
    }
    return a;
}

// ---------------------------------------------------------------- kNN top-k
// one thread per point; query row cached in LDS; score = 2<xn,xm> - |xm|^2
__global__ void topk_kernel(const float* __restrict__ F, int rowStride, int C,
                            int* __restrict__ idxOut) {
    extern __shared__ float sXn[];                     // blockDim.x * C
    const int tid = threadIdx.x;
    const int row = blockIdx.x * blockDim.x + tid;     // global point index
    const int b = row / NPTS, n = row % NPTS;
    const float* Fb = F + (size_t)b * NPTS * rowStride;
    float* xn = sXn + (size_t)tid * C;
    for (int c = 0; c < C; ++c) xn[c] = Fb[(size_t)n * rowStride + c];

    float sc[KNN]; int id[KNN];
#pragma unroll
    for (int i = 0; i < KNN; ++i) { sc[i] = -__builtin_inff(); id[i] = 0; }

    for (int m = 0; m < NPTS; ++m) {
        const float* xm = Fb + (size_t)m * rowStride;
        __builtin_prefetch(xm + rowStride, 0, 0);      // global_prefetch_b8
        float dot = 0.f, sq = 0.f;
        for (int c = 0; c < C; ++c) { float v = xm[c]; dot += xn[c] * v; sq += v * v; }
        const float s = 2.f * dot - sq;
        if (s > sc[KNN - 1]) {
            int j = KNN - 1;
            while (j > 0 && sc[j - 1] < s) { sc[j] = sc[j - 1]; id[j] = id[j - 1]; --j; }
            sc[j] = s; id[j] = m;
        }
    }
    for (int i = 0; i < KNN; ++i) idxOut[(size_t)row * KNN + i] = id[i];
}

// ---------------------------------------------------------------- weight prepack
// w: [Cout, Cin] row-major; produce B-fragments (wT) in lane-contiguous order:
// frag(kt,nt): wpack[((kt*NT+nt)*32 + lane)*16 + e]
__global__ void prepack_w_kernel(const float* __restrict__ w, __bf16* __restrict__ wpack,
                                 int Cin, int Cout, int Kpad) {
    const int t = blockIdx.x * blockDim.x + threadIdx.x;
    const int NT = Cout >> 4;
    const int total = (Kpad >> 5) * NT * 32;
    if (t >= total) return;
    const int lane = t & 31;
    const int tile = t >> 5;
    const int kt = tile / NT, nt = tile % NT;
    const int ncol = nt * 16 + (lane & 15);
    const int half = lane >> 4;
    __bf16* dst = wpack + (size_t)t * 16;
#pragma unroll
    for (int v = 0; v < 8; ++v) {
        const int kbase = kt * 32 + (v >> 2) * 16 + half * 8 + (v & 3) * 2;
#pragma unroll
        for (int pos = 0; pos < 2; ++pos) {
            const int k = kbase + pos;
            float val = (k < Cin) ? w[(size_t)ncol * Cin + k] : 0.f;
            dst[2 * v + pos] = (__bf16)val;
        }
    }
}

// ---------------------------------------------------------------- fused edge conv
// gather edge features -> bf16 LDS -> WMMA -> epilogue: max over k to global,
// per-channel sum / sumsq atomics for BN batch statistics.
__global__ void edgeconv_kernel(const float* __restrict__ F, int rowStride, int C,
                                const int* __restrict__ idx,
                                const __bf16* __restrict__ wpack,
                                float* __restrict__ outBase, int outStride, int Cout,
                                float* __restrict__ sumBuf, float* __restrict__ sumsqBuf,
                                int Kpad) {
    extern __shared__ char smem[];
    const int ROWS = TNPTS * KNN;                       // 80
    float*  ldsY   = (float*)smem;                      // ROWS x 16 (one col-tile)
    int*    ldsIdx = (int*)(ldsY + ROWS * 16);          // ROWS
    __bf16* ldsA   = (__bf16*)(ldsIdx + ROWS);          // ROWS x Kpad

    const int tid = threadIdx.x, wave = tid >> 5, lane = tid & 31;
    const int Cin = 2 * C;
    const int grp0 = blockIdx.x * TNPTS;                // first global row (b*N+n)
    const int b = grp0 / NPTS;
    const float* Fb = F + (size_t)b * NPTS * rowStride;

    for (int r = tid; r < ROWS; r += blockDim.x)
        ldsIdx[r] = idx[(size_t)(grp0 + r / KNN) * KNN + (r % KNN)];
    __syncthreads();

    // stage A tile: feat = [x_nbr - x_n | x_n], zero pad to Kpad
    const int totalA = ROWS * Kpad;
    for (int e = tid; e < totalA; e += blockDim.x) {
        const int r = e / Kpad, c = e % Kpad;
        const int p = r / KNN;
        const int n = (grp0 + p) % NPTS;
        float v = 0.f;
        if (c < Cin) {
            if (c < C) {
                const int nb = ldsIdx[r];
                v = Fb[(size_t)nb * rowStride + c] - Fb[(size_t)n * rowStride + c];
            } else {
                v = Fb[(size_t)n * rowStride + (c - C)];
            }
        }
        ldsA[(size_t)r * Kpad + c] = (__bf16)v;
    }
    __syncthreads();

    const int NT = Cout >> 4;
    const int KT = Kpad >> 5;
    const int MT = ROWS / 16;                           // 5
    for (int nt = 0; nt < NT; ++nt) {
        for (int mt = wave; mt < MT; mt += 4) {         // wave-uniform: EXEC all ones
            v8f acc = {};
            for (int kt = 0; kt < KT; ++kt) {
                v16bf a = load_a_frag(ldsA, Kpad, mt * 16, kt * 32, lane);
                v16bf bf = *(const v16bf*)(wpack + ((size_t)(kt * NT + nt) * 32 + lane) * 16);
                acc = __builtin_amdgcn_wmma_f32_16x16x32_bf16(
                          false, a, false, bf, (short)0, acc, false, false);
            }
            const int col = lane & 15, rbase = (lane >> 4) * 8;
#pragma unroll
            for (int v = 0; v < 8; ++v)
                ldsY[(mt * 16 + rbase + v) * 16 + col] = acc[v];
        }
        __syncthreads();
        // max over k for this 16-channel tile
        if (tid < TNPTS * 16) {
            const int p = tid >> 4, oc = tid & 15;
            float mx = -__builtin_inff();
            for (int kk = 0; kk < KNN; ++kk)
                mx = fmaxf(mx, ldsY[(p * KNN + kk) * 16 + oc]);
            outBase[(size_t)(grp0 + p) * outStride + nt * 16 + oc] = mx;
        }
        // batch stats over all 80 real rows
        if (tid < 16) {
            const int oc = tid;
            float sm = 0.f, sq = 0.f;
            for (int r = 0; r < ROWS; ++r) {
                const float y = ldsY[r * 16 + oc];
                sm += y; sq += y * y;
            }
            atomicAdd(&sumBuf[nt * 16 + oc], sm);
            atomicAdd(&sumsqBuf[nt * 16 + oc], sq);
        }
        __syncthreads();
    }
}

// ---------------------------------------------------------------- conv5 (512->512) fused
__global__ void conv5_kernel(const float* __restrict__ cat,
                             const __bf16* __restrict__ wpack,
                             float* __restrict__ y5max,
                             float* __restrict__ sumBuf, float* __restrict__ sumsqBuf) {
    extern __shared__ char smem[];
    float*  ldsY = (float*)smem;                        // C5ROWS x 16
    __bf16* ldsA = (__bf16*)(ldsY + C5ROWS * 16);       // C5ROWS x 512

    const int tid = threadIdx.x, wave = tid >> 5, lane = tid & 31;
    const int row0 = blockIdx.x * C5ROWS;
    const int b = row0 / NPTS;

    for (int e = tid; e < C5ROWS * 512; e += blockDim.x)
        ldsA[e] = (__bf16)cat[(size_t)row0 * 512 + e];
    __syncthreads();

    const int NT = 32, KT = 16, MT = C5ROWS / 16;
    for (int nt = 0; nt < NT; ++nt) {
        for (int mt = wave; mt < MT; mt += 4) {
            v8f acc = {};
            for (int kt = 0; kt < KT; ++kt) {
                v16bf a = load_a_frag(ldsA, 512, mt * 16, kt * 32, lane);
                v16bf bf = *(const v16bf*)(wpack + ((size_t)(kt * NT + nt) * 32 + lane) * 16);
                acc = __builtin_amdgcn_wmma_f32_16x16x32_bf16(
                          false, a, false, bf, (short)0, acc, false, false);
            }
            const int col = lane & 15, rbase = (lane >> 4) * 8;
#pragma unroll
            for (int v = 0; v < 8; ++v)
                ldsY[(mt * 16 + rbase + v) * 16 + col] = acc[v];
        }
        __syncthreads();
        if (tid < 16) {
            const int oc = tid;
            float mx = -__builtin_inff(), sm = 0.f, sq = 0.f;
            for (int r = 0; r < C5ROWS; ++r) {
                const float y = ldsY[r * 16 + oc];
                mx = fmaxf(mx, y); sm += y; sq += y * y;
            }
            atomicMaxFloat(&y5max[(size_t)b * 512 + nt * 16 + oc], mx);
            atomicAdd(&sumBuf[nt * 16 + oc], sm);
            atomicAdd(&sumsqBuf[nt * 16 + oc], sq);
        }
        __syncthreads();
    }
}

// ---------------------------------------------------------------- BN helpers
__global__ void bn_prepare_kernel(const float* __restrict__ sum, const float* __restrict__ sumsq,
                                  const float* __restrict__ gamma, const float* __restrict__ beta,
                                  float* __restrict__ scale, float* __restrict__ shift,
                                  int Cout, float invCount) {
    const int o = blockIdx.x * blockDim.x + threadIdx.x;
    if (o >= Cout) return;
    const float mu  = sum[o] * invCount;
    const float var = sumsq[o] * invCount - mu * mu;
    const float sc  = gamma[o] * rsqrtf(var + EPSBN);
    scale[o] = sc;
    shift[o] = beta[o] - mu * sc;
}

__global__ void bn_apply_kernel(float* __restrict__ data, int stride, int Cout,
                                const float* __restrict__ scale, const float* __restrict__ shift,
                                int rows) {
    const int t = blockIdx.x * blockDim.x + threadIdx.x;
    if (t >= rows * Cout) return;
    const int r = t / Cout, o = t % Cout;
    float* p = data + (size_t)r * stride + o;
    float y = *p * scale[o] + shift[o];
    *p = (y > 0.f) ? y : SLOPE * y;
}

__global__ void fill_neg_inf_kernel(float* __restrict__ p, int n) {
    const int t = blockIdx.x * blockDim.x + threadIdx.x;
    if (t < n) p[t] = -__builtin_inff();
}

// ---------------------------------------------------------------- final embedding
__global__ void final_kernel(const float* __restrict__ y5max,
                             const float* __restrict__ scale, const float* __restrict__ shift,
                             const float* __restrict__ we, float* __restrict__ out, int Fdim) {
    const int t = blockIdx.x * blockDim.x + threadIdx.x;
    if (t >= BATCH * Fdim) return;
    const int b = t / Fdim, f = t % Fdim;
    const float* yb = y5max + (size_t)b * 512;
    const float* wf = we + (size_t)f * 512;
    float acc = 0.f;
    for (int o = 0; o < 512; ++o) {
        float y = yb[o] * scale[o] + shift[o];
        y = (y > 0.f) ? y : SLOPE * y;
        acc += y * wf[o];
    }
    out[t] = acc;
}

// ---------------------------------------------------------------- host
extern "C" void kernel_launch(void* const* d_in, const int* in_sizes, int n_in,
                              void* d_out, int out_size, void* d_ws, size_t ws_size,
                              hipStream_t stream) {
    const float* x  = (const float*)d_in[0];
    const float* W[5]  = { (const float*)d_in[1],  (const float*)d_in[4],
                           (const float*)d_in[7],  (const float*)d_in[10],
                           (const float*)d_in[13] };
    const float* G[5]  = { (const float*)d_in[2],  (const float*)d_in[5],
                           (const float*)d_in[8],  (const float*)d_in[11],
                           (const float*)d_in[14] };
    const float* Bt[5] = { (const float*)d_in[3],  (const float*)d_in[6],
                           (const float*)d_in[9],  (const float*)d_in[12],
                           (const float*)d_in[15] };
    const float* we = (const float*)d_in[16];
    (void)n_in; (void)in_sizes; (void)ws_size;

    char* ws = (char*)d_ws;
    size_t off = 0;
    float*  cat   = (float*)(ws + off); off += (size_t)BATCH * NPTS * 512 * sizeof(float);
    int*    idx   = (int*)(ws + off);   off += (size_t)BATCH * NPTS * KNN * sizeof(int);
    __bf16* wpack = (__bf16*)(ws + off); off += (size_t)512 * 512 * sizeof(__bf16);
    float*  stats = (float*)(ws + off); off += (size_t)5 * 1024 * sizeof(float);
    float*  scsh  = (float*)(ws + off); off += (size_t)1024 * sizeof(float);
    float*  y5max = (float*)(ws + off); off += (size_t)BATCH * 512 * sizeof(float);

    hipMemsetAsync(stats, 0, 5 * 1024 * sizeof(float), stream);
    fill_neg_inf_kernel<<<(BATCH * 512 + 255) / 256, 256, 0, stream>>>(y5max, BATCH * 512);

    struct BlkCfg { const float* in; int inStride, C, Cout, outOff;
                    const float *w, *g, *bt; };
    const BlkCfg blk[4] = {
        { x,         3,   3,   64,  0,   W[0], G[0], Bt[0] },
        { cat + 0,   512, 64,  64,  64,  W[1], G[1], Bt[1] },
        { cat + 64,  512, 64,  128, 128, W[2], G[2], Bt[2] },
        { cat + 128, 512, 128, 256, 256, W[3], G[3], Bt[3] },
    };

    const int ROWS = TNPTS * KNN;
    for (int i = 0; i < 4; ++i) {
        const BlkCfg& c = blk[i];
        const int Cin  = 2 * c.C;
        const int Kpad = (Cin + 31) & ~31;

        // kNN top-k
        topk_kernel<<<(BATCH * NPTS) / 64, 64, (size_t)64 * c.C * sizeof(float), stream>>>(
            c.in, c.inStride, c.C, idx);

        // pack weights into B-fragment order
        {
            const int tot = (Kpad / 32) * (c.Cout / 16) * 32;
            prepack_w_kernel<<<(tot + 127) / 128, 128, 0, stream>>>(
                c.w, wpack, Cin, c.Cout, Kpad);
        }

        // fused edge conv + max-over-k + BN stats
        {
            const size_t lds = (size_t)ROWS * 16 * sizeof(float)
                             + (size_t)ROWS * sizeof(int)
                             + (size_t)ROWS * Kpad * sizeof(__bf16);
            edgeconv_kernel<<<(BATCH * NPTS) / TNPTS, 128, lds, stream>>>(
                c.in, c.inStride, c.C, idx, wpack,
                cat + c.outOff, 512, c.Cout,
                stats + i * 1024, stats + i * 1024 + 512, Kpad);
        }

        // BN affine + leaky relu applied in place to the pooled features
        bn_prepare_kernel<<<(c.Cout + 255) / 256, 256, 0, stream>>>(
            stats + i * 1024, stats + i * 1024 + 512, c.g, c.bt,
            scsh, scsh + 512, c.Cout, 1.0f / ((float)BATCH * NPTS * KNN));
        {
            const int tot = BATCH * NPTS * c.Cout;
            bn_apply_kernel<<<(tot + 255) / 256, 256, 0, stream>>>(
                cat + c.outOff, 512, c.Cout, scsh, scsh + 512, BATCH * NPTS);
        }
    }

    // conv5: 512 -> 512, fused global-max-pool + BN stats
    {
        const int tot = (512 / 32) * (512 / 16) * 32;
        prepack_w_kernel<<<(tot + 127) / 128, 128, 0, stream>>>(W[4], wpack, 512, 512, 512);
        const size_t lds = (size_t)C5ROWS * 16 * sizeof(float)
                         + (size_t)C5ROWS * 512 * sizeof(__bf16);
        conv5_kernel<<<(BATCH * NPTS) / C5ROWS, 128, lds, stream>>>(
            cat, wpack, y5max, stats + 4 * 1024, stats + 4 * 1024 + 512);
        bn_prepare_kernel<<<(512 + 255) / 256, 256, 0, stream>>>(
            stats + 4 * 1024, stats + 4 * 1024 + 512, G[4], Bt[4],
            scsh, scsh + 512, 512, 1.0f / ((float)BATCH * NPTS));
    }

    // final embedding
    {
        const int Fdim = out_size / BATCH;
        const int tot = BATCH * Fdim;
        final_kernel<<<(tot + 255) / 256, 256, 0, stream>>>(
            y5max, scsh, scsh + 512, we, (float*)d_out, Fdim);
    }
}
